// ImageBlendSynth_26448408609231
// MI455X (gfx1250) — matrix operand order repaired and checked by reading, not compiled, not run on hardware
//
#include <hip/hip_runtime.h>
#include <hip/hip_bf16.h>

#define NFFT   1024
#define HOPS   256
#define NB     513
#define NMEL   128
#define NFRM   344
#define NBAT   8
#define BT     (NBAT*NFRM)   /* 2752 */
#define KSP    1056          /* 1026 padded to mult of 32 */
#define NSP    1088          /* 1026 padded to mult of 64 */
#define OUTLEN 88832         /* NFFT + HOPS*(NFRM-1) */
#define INVLEN 87808         /* OUTLEN - NFFT */
#define NSAMP  88200
#define PI2F   6.283185307179586f

typedef __attribute__((ext_vector_type(16))) _Float16 v16h;
typedef __attribute__((ext_vector_type(8)))  _Float16 v8h;
typedef __attribute__((ext_vector_type(8)))  float    v8f;

__device__ __forceinline__ v16h frag16(const _Float16* p0, const _Float16* p1) {
    v16h r;
    ((v8h*)&r)[0] = *(const v8h*)p0;
    ((v8h*)&r)[1] = *(const v8h*)p1;
    return r;
}

// ---------------------------------------------------------------------------
// WMMA GEMM:  C[M x N] (f32) = A[M x K] (f16, row-major) * Bt[N x K]^T (f16)
// Bt is B stored transposed (row n = column n of B), so both operands load as
// contiguous 16B chunks in one shared contiguous k order (k = 16*hi + s) for A and B.
// Block = 128 threads (4 waves); wave computes a 16(M) x 64(N) strip.
// M multiple of 64, N multiple of 64, K multiple of 32 (zero-padded).
// ---------------------------------------------------------------------------
__global__ __launch_bounds__(128)
void k_wmma_gemm(const _Float16* __restrict__ A, const _Float16* __restrict__ Bt,
                 float* __restrict__ C, int M, int N, int K)
{
    const int wave = threadIdx.x >> 5;
    const int lane = threadIdx.x & 31;
    const int r    = lane & 15;
    const int hi   = lane >> 4;
    const int m0   = (blockIdx.y * 4 + wave) * 16;
    const int n0   = blockIdx.x * 64;
    if (m0 >= M) return;                       // wave-uniform: EXEC stays all-1s

    v8f acc0 = {}, acc1 = {}, acc2 = {}, acc3 = {};

    const _Float16* Arow = A + (size_t)(m0 + r) * K;
    const _Float16* B0   = Bt + (size_t)(n0 +  0 + r) * K + hi * 16;
    const _Float16* B1   = Bt + (size_t)(n0 + 16 + r) * K + hi * 16;
    const _Float16* B2   = Bt + (size_t)(n0 + 32 + r) * K + hi * 16;
    const _Float16* B3   = Bt + (size_t)(n0 + 48 + r) * K + hi * 16;

    for (int k = 0; k < K; k += 32) {
        // A fragment: lane r rows m0+r, K-pattern {k+hi*16 .. +16} contiguous
        v16h af = frag16(Arow + k + hi * 16, Arow + k + hi * 16 + 8);
        // B fragments: lane holds one column, 16 contiguous K values
        v16h b0 = frag16(B0 + k, B0 + k + 8);
        v16h b1 = frag16(B1 + k, B1 + k + 8);
        v16h b2 = frag16(B2 + k, B2 + k + 8);
        v16h b3 = frag16(B3 + k, B3 + k + 8);
        acc0 = __builtin_amdgcn_wmma_f32_16x16x32_f16(false, af, false, b0, (short)0, acc0, false, false);
        acc1 = __builtin_amdgcn_wmma_f32_16x16x32_f16(false, af, false, b1, (short)0, acc1, false, false);
        acc2 = __builtin_amdgcn_wmma_f32_16x16x32_f16(false, af, false, b2, (short)0, acc2, false, false);
        acc3 = __builtin_amdgcn_wmma_f32_16x16x32_f16(false, af, false, b3, (short)0, acc3, false, false);
    }

    // C/D layout: VGPR v -> row m0 + v + hi*8, lane r -> column n0 + 16*j + r
#pragma unroll
    for (int v = 0; v < 8; ++v) {
        size_t row = (size_t)(m0 + v + hi * 8) * N;
        C[row + n0 +  0 + r] = acc0[v];
        C[row + n0 + 16 + r] = acc1[v];
        C[row + n0 + 32 + r] = acc2[v];
        C[row + n0 + 48 + r] = acc3[v];
    }
}

// --------------------------- init kernels ----------------------------------
__global__ void k_win_env(float* __restrict__ window, float* __restrict__ env_inv)
{
    int s = blockIdx.x * blockDim.x + threadIdx.x;     // 0..OUTLEN-1
    if (s >= OUTLEN) return;
    if (s < NFFT) window[s] = 0.5f - 0.5f * cosf(PI2F * (float)s / (float)NFFT);
    int thi = s >> 8;  if (thi > NFRM - 1) thi = NFRM - 1;
    int tlo = (s >> 8) - 3;  if (tlo < 0) tlo = 0;
    float e = 0.f;
    for (int t = tlo; t <= thi; ++t) {
        int n = s - (t << 8);
        float wn = 0.5f - 0.5f * cosf(PI2F * (float)n / (float)NFFT);
        e += wn * wn;
    }
    env_inv[s] = 1.f / fmaxf(e, 1e-11f);
}

// Forward rDFT, transposed: DfwdT[j][n], j<513 -> cos, 513..1025 -> -sin, rest 0
__global__ void k_gen_dfwd(_Float16* __restrict__ DfwdT)
{
    int idx = blockIdx.x * blockDim.x + threadIdx.x;   // NSP*NFFT
    if (idx >= NSP * NFFT) return;
    int n = idx & (NFFT - 1);
    int j = idx >> 10;
    float v = 0.f;
    if (j < NB) {
        int ph = (j * n) & (NFFT - 1);
        v = cosf((float)ph * (PI2F / NFFT));
    } else if (j < 2 * NB) {
        int k = j - NB;
        int ph = (k * n) & (NFFT - 1);
        v = -sinf((float)ph * (PI2F / NFFT));
    }
    DfwdT[idx] = (_Float16)v;
}

// Inverse rDFT (window folded in), transposed: DinvT[n][j]
__global__ void k_gen_dinv(_Float16* __restrict__ DinvT)
{
    int idx = blockIdx.x * blockDim.x + threadIdx.x;   // NFFT*KSP
    if (idx >= NFFT * KSP) return;
    int j = idx % KSP;
    int n = idx / KSP;
    float wn = 0.5f - 0.5f * cosf(PI2F * (float)n / (float)NFFT);
    float v = 0.f;
    if (j <= 512) {                                    // Re coefficients
        float wk = (j == 0 || j == 512) ? 1.f : 2.f;
        int ph = (j * n) & (NFFT - 1);
        v = wk * (1.f / NFFT) * cosf((float)ph * (PI2F / NFFT));
    } else if (j < 2 * NB) {                           // Im coefficients
        int k = j - NB;
        if (k != 0 && k != 512) {
            int ph = (k * n) & (NFFT - 1);
            v = -(2.f / NFFT) * sinf((float)ph * (PI2F / NFFT));
        }
    }
    DinvT[idx] = (_Float16)(v * wn);
}

__global__ void k_tgt_red(const float* __restrict__ tgt, float* __restrict__ red)
{
    __shared__ float smn[256], smx[256];
    float mn = 3.4e38f, mx = -3.4e38f;
    for (int i = threadIdx.x; i < NMEL * NFRM; i += 256) {
        float v = tgt[i]; mn = fminf(mn, v); mx = fmaxf(mx, v);
    }
    smn[threadIdx.x] = mn; smx[threadIdx.x] = mx; __syncthreads();
    for (int s = 128; s > 0; s >>= 1) {
        if (threadIdx.x < s) {
            smn[threadIdx.x] = fminf(smn[threadIdx.x], smn[threadIdx.x + s]);
            smx[threadIdx.x] = fmaxf(smx[threadIdx.x], smx[threadIdx.x + s]);
        }
        __syncthreads();
    }
    if (threadIdx.x == 0) { red[0] = smn[0]; red[1] = smx[0]; }
}

// mspec factorizes: mspec[b][f][t] = G[b][f] * E[b][t]; max = maxG*maxE.
// mbuf per-b stride 512: [0..127]=G, [128..471]=E, [472]=scale, [473..476]=blend params
__global__ void k_mel_fac(const float* __restrict__ params, float* __restrict__ mbuf)
{
    int b = blockIdx.x, t = threadIdx.x;               // 128 threads
    const float lo[11] = {0,0,0,0,0.05f,3.0f,0.3f,1.0f,0.0f,0.6f,0.7f};
    const float hiR[11]= {1,1,1,1,0.25f,12.0f,0.8f,4.0f,0.3f,1.0f,1.0f};
    float p[11];
    for (int i = 0; i < 11; ++i) p[i] = params[b*11 + i] * (hiR[i] - lo[i]) + lo[i];
    float fund = floorf(p[4] * (float)NMEL);
    fund = fminf(fmaxf(fund, 2.f), (float)(NMEL / 4));
    float nh = floorf(p[5]), decay = p[6], bw = p[7], attack = p[8], sustain = p[9], inten = p[10];

    __shared__ float red[128];
    float G = 0.f;
    {
        float mel = (float)t;
        for (int h = 1; h <= 12; ++h) {
            float hf = (float)h;
            float hbin = fund * hf;
            bool valid = (hf <= nh) && (hbin < (float)NMEL);
            float amp = valid ? inten / powf(hf, decay) : 0.f;
            float d = (mel - hbin) / bw;
            G += amp * expf(-0.5f * d * d);
        }
        mbuf[b*512 + t] = G;
    }
    red[t] = G; __syncthreads();
    for (int s = 64; s > 0; s >>= 1) { if (t < s) red[t] = fmaxf(red[t], red[t + s]); __syncthreads(); }
    float maxG = red[0]; __syncthreads();

    float mloc = 0.f;
    for (int tt = t; tt < NFRM; tt += 128) {
        float tn = (float)tt / (float)NFRM;
        float e = fminf(fmaxf(tn / (attack + 0.01f), 0.f), 1.f) * sustain;
        mbuf[b*512 + 128 + tt] = e;
        mloc = fmaxf(mloc, e);
    }
    red[t] = mloc; __syncthreads();
    for (int s = 64; s > 0; s >>= 1) { if (t < s) red[t] = fmaxf(red[t], red[t + s]); __syncthreads(); }
    if (t == 0) {
        float mx = maxG * red[0];
        mbuf[b*512 + 472] = (mx > 0.f) ? 1.f / fmaxf(mx, 1e-30f) : 1.f;
        mbuf[b*512 + 473] = p[0];
        mbuf[b*512 + 474] = p[1];
        mbuf[b*512 + 475] = p[2];
        mbuf[b*512 + 476] = p[3];
    }
}

__global__ void k_mag(const float* __restrict__ tgt, const float* __restrict__ red,
                      const float* __restrict__ mbuf, float* __restrict__ mag)
{
    int idx = blockIdx.x * blockDim.x + threadIdx.x;   // NBAT*NFRM*NB
    if (idx >= NBAT * NFRM * NB) return;
    int k = idx % NB;
    int t = (idx / NB) % NFRM;
    int b = idx / (NB * NFRM);
    float tmin = red[0], tmax = red[1];
    float tsc = 1.f / (tmax - tmin + 1e-8f);
    const float* mb = mbuf + b * 512;
    float scale = mb[472];
    float bg = mb[473], bl = mb[474], bm = mb[475], bh = mb[476];
    float E = mb[128 + t];
    float src = ((float)k + 0.5f) * ((float)NMEL / (float)NB) - 0.5f;
    src = fminf(fmaxf(src, 0.f), (float)(NMEL - 1));
    int i0 = (int)floorf(src);
    int i1 = (i0 + 1 < NMEL - 1) ? i0 + 1 : NMEL - 1;
    float w = src - (float)i0;
    float sp[2]; int ii[2] = {i0, i1};
#pragma unroll
    for (int jj = 0; jj < 2; ++jj) {
        int f = ii[jj];
        float fb = (float)f * (1.f / (float)(NMEL - 1));
        float add = (fb < 0.33f) ? (bl - 0.5f) * 0.3f
                  : (fb < 0.66f) ? (bm - 0.5f) * 0.3f
                                 : (bh - 0.5f) * 0.3f;
        float blend = fminf(fmaxf(bg + add, 0.f), 1.f);
        float ms = mb[f] * E * scale;
        float tn = (tgt[f * NFRM + t] - tmin) * tsc;
        float bs = blend * ms + (1.f - blend) * tn;
        sp[jj] = bs * bs * 100.f;
    }
    float lin = sp[0] * (1.f - w) + sp[1] * w;
    mag[((size_t)(b * NFRM + t)) * NB + k] = sqrtf(lin);
}

__global__ void k_sh_init(const float* __restrict__ mag, _Float16* __restrict__ SH)
{
    int idx = blockIdx.x * blockDim.x + threadIdx.x;   // BT*KSP
    if (idx >= BT * KSP) return;
    int j = idx % KSP, bt = idx / KSP;
    SH[idx] = (j < NB) ? (_Float16)mag[(size_t)bt * NB + j] : (_Float16)0.f;
}

__global__ void k_zero_f32(float* __restrict__ p, int n)
{
    int idx = blockIdx.x * blockDim.x + threadIdx.x;
    if (idx < n) p[idx] = 0.f;
}

// --------------------------- loop kernels ----------------------------------
__global__ void k_ola(const float* __restrict__ frames, const float* __restrict__ env_inv,
                      float* __restrict__ sig)
{
    int idx = blockIdx.x * blockDim.x + threadIdx.x;   // NBAT*OUTLEN
    if (idx >= NBAT * OUTLEN) return;
    int s = idx % OUTLEN, b = idx / OUTLEN;
    int thi = s >> 8;  if (thi > NFRM - 1) thi = NFRM - 1;
    int tlo = (s >> 8) - 3;  if (tlo < 0) tlo = 0;
    float acc = 0.f;
    for (int t = tlo; t <= thi; ++t)
        acc += frames[((size_t)(b * NFRM + t)) * NFFT + (s - (t << 8))];
    sig[idx] = acc * env_inv[s];
}

__global__ void k_reframe(const float* __restrict__ sig, const float* __restrict__ window,
                          _Float16* __restrict__ framesH)
{
    int idx = blockIdx.x * blockDim.x + threadIdx.x;   // NBAT*NFRM*NFFT
    if (idx >= NBAT * NFRM * NFFT) return;
    int n = idx & (NFFT - 1);
    int t = (idx >> 10) % NFRM;
    int b = idx / (NFRM * NFFT);
    int m = (t << 8) + n - NFFT / 2;                   // reflect into [0, INVLEN)
    if (m < 0) m = -m;
    if (m >= INVLEN) m = 2 * (INVLEN - 1) - m;
    float v = sig[(size_t)b * OUTLEN + m + NFFT / 2] * window[n];
    framesH[idx] = (_Float16)v;
}

__global__ void k_phase(const float* __restrict__ cur, const float* __restrict__ prev,
                        const float* __restrict__ mag, _Float16* __restrict__ SH)
{
    int idx = blockIdx.x * blockDim.x + threadIdx.x;   // BT*NB
    if (idx >= BT * NB) return;
    int k = idx % NB, bt = idx / NB;
    const float m = 0.99f / 1.99f;                     // MOMENTUM/(1+MOMENTUM)
    size_t row = (size_t)bt * NSP;
    float ar = cur[row + k]      - m * prev[row + k];
    float ai = cur[row + NB + k] - m * prev[row + NB + k];
    float nrm = sqrtf(ar * ar + ai * ai) + 1e-16f;
    float mg = mag[(size_t)bt * NB + k];
    size_t srow = (size_t)bt * KSP;
    SH[srow + k]      = (_Float16)(mg * ar / nrm);
    SH[srow + NB + k] = (_Float16)(mg * ai / nrm);
}

__global__ void k_final(const float* __restrict__ sig, float* __restrict__ out)
{
    int b = blockIdx.x, t = threadIdx.x;               // 8 blocks x 256
    __shared__ float red[256];
    float mx = 0.f;
    for (int s = t; s < NSAMP; s += 256) {
        float v = (s < INVLEN) ? sig[(size_t)b * OUTLEN + NFFT / 2 + s] : 0.f;
        mx = fmaxf(mx, fabsf(v));
    }
    red[t] = mx; __syncthreads();
    for (int s = 128; s > 0; s >>= 1) { if (t < s) red[t] = fmaxf(red[t], red[t + s]); __syncthreads(); }
    float inv = 0.9f / fmaxf(red[0], 1e-8f);
    for (int s = t; s < NSAMP; s += 256) {
        float v = (s < INVLEN) ? sig[(size_t)b * OUTLEN + NFFT / 2 + s] : 0.f;
        out[(size_t)b * NSAMP + s] = v * inv;
    }
}

// --------------------------- host orchestration ----------------------------
extern "C" void kernel_launch(void* const* d_in, const int* in_sizes, int n_in,
                              void* d_out, int out_size, void* d_ws, size_t ws_size,
                              hipStream_t stream)
{
    const float* params = (const float*)d_in[0];       // (8,11) f32
    const float* tgt    = (const float*)d_in[1];       // (1,128,344) f32
    float*       out    = (float*)d_out;               // (8,88200) f32

    char* base = (char*)d_ws;
    size_t off = 0;
    auto take = [&](size_t bytes) -> char* {
        char* p = base + off;
        off = (off + bytes + 255) & ~(size_t)255;
        return p;
    };
    float*    window  = (float*)   take((size_t)NFFT * 4);
    float*    env_inv = (float*)   take((size_t)OUTLEN * 4);
    _Float16* DfwdT   = (_Float16*)take((size_t)NSP * NFFT * 2);
    _Float16* DinvT   = (_Float16*)take((size_t)NFFT * KSP * 2);
    float*    redbuf  = (float*)   take(64 * 4);
    float*    mbuf    = (float*)   take((size_t)NBAT * 512 * 4);
    float*    mag     = (float*)   take((size_t)BT * NB * 4);
    _Float16* SH      = (_Float16*)take((size_t)BT * KSP * 2);
    float*    specA   = (float*)   take((size_t)BT * NSP * 4);
    float*    specB   = (float*)   take((size_t)BT * NSP * 4);
    float*    frames  = (float*)   take((size_t)BT * NFFT * 4);
    _Float16* framesH = (_Float16*)take((size_t)BT * NFFT * 2);
    float*    sig     = (float*)   take((size_t)NBAT * OUTLEN * 4);
    (void)ws_size; (void)in_sizes; (void)n_in; (void)out_size;

    // ---- init ----
    k_win_env <<<(OUTLEN + 255) / 256, 256, 0, stream>>>(window, env_inv);
    k_gen_dfwd<<<(NSP * NFFT) / 256, 256, 0, stream>>>(DfwdT);
    k_gen_dinv<<<(NFFT * KSP) / 256, 256, 0, stream>>>(DinvT);
    k_tgt_red <<<1, 256, 0, stream>>>(tgt, redbuf);
    k_mel_fac <<<NBAT, 128, 0, stream>>>(params, mbuf);
    k_mag     <<<(NBAT * NFRM * NB + 255) / 256, 256, 0, stream>>>(tgt, redbuf, mbuf, mag);
    k_sh_init <<<(BT * KSP) / 256, 256, 0, stream>>>(mag, SH);
    k_zero_f32<<<(BT * NSP) / 256, 256, 0, stream>>>(specA, BT * NSP);

    // ---- Griffin-Lim loop: 32 iterations ----
    float* prev = specA;   // tprev (zeroed)
    float* cur  = specB;
    const dim3 gInv(NFFT / 64, (BT / 16) / 4);         // (16, 43)
    const dim3 gFwd(NSP  / 64, (BT / 16) / 4);         // (17, 43)
    for (int it = 0; it < 32; ++it) {
        k_wmma_gemm<<<gInv, 128, 0, stream>>>(SH, DinvT, frames, BT, NFFT, KSP);
        k_ola      <<<(NBAT * OUTLEN) / 256, 256, 0, stream>>>(frames, env_inv, sig);
        k_reframe  <<<(NBAT * NFRM * NFFT) / 256, 256, 0, stream>>>(sig, window, framesH);
        k_wmma_gemm<<<gFwd, 128, 0, stream>>>(framesH, DfwdT, cur, BT, NSP, NFFT);
        k_phase    <<<(BT * NB + 255) / 256, 256, 0, stream>>>(cur, prev, mag, SH);
        float* tmp = prev; prev = cur; cur = tmp;
    }

    // ---- final istft + normalize ----
    k_wmma_gemm<<<gInv, 128, 0, stream>>>(SH, DinvT, frames, BT, NFFT, KSP);
    k_ola      <<<(NBAT * OUTLEN) / 256, 256, 0, stream>>>(frames, env_inv, sig);
    k_final    <<<NBAT, 256, 0, stream>>>(sig, out);
}
